// Model_76184129896453
// MI455X (gfx1250) — compile-verified, matrix operand order checked by reading, not run on hardware
//
#include <hip/hip_runtime.h>
#include <hip/hip_bf16.h>

typedef __attribute__((ext_vector_type(16))) _Float16 v16h;
typedef __attribute__((ext_vector_type(8)))  float    v8f;

#define TILE   64
#define STEP   32
#define MD     8
#define DGRID  17          // 2*MD+1
#define WIN    80          // TILE + 2*MD
#define H_IMG  2048
#define W_IMG  2048
#define TGRID  63          // (2048-64)/32 + 1

#define SS_STRIDE  72      // src tile LDS row stride (halves)
#define SW_STRIDE  96      // window LDS row stride (halves), covers k+col+1 <= 94
#define SAT_STRIDE 84      // SAT LDS row stride (floats)

__global__ __launch_bounds__(256)
void bm_cost_kernel(const float* __restrict__ src,
                    const float* __restrict__ tgt,
                    float* __restrict__ tile_out /* [TGRID*TGRID][2] = (dx,dy) */)
{
    __shared__ _Float16 sS[TILE + 1][SS_STRIDE];     // row 64 = zero row (OOB clamp)
    __shared__ _Float16 sW[WIN][SW_STRIDE];
    __shared__ float    sSAT[WIN + 1][SAT_STRIDE];   // SAT of w^2, 81x81 used
    __shared__ float    sCorr[DGRID][20];
    __shared__ float    sCost[DGRID * DGRID];
    __shared__ float    sRedF[256];
    __shared__ int      sRedI[256];

    const int tx   = blockIdx.x;
    const int ty   = blockIdx.y;
    const int tid  = threadIdx.x;
    const int lane = tid & 31;
    const int wave = tid >> 5;

    const int ys = ty * STEP;
    const int xs = tx * STEP;

    // ---- phase 1: load src tile (f32 -> f16), local src_sq / overlap; zero sW, sCorr, zero-row ----
    float mySq = 0.f;
    int   myOv = 0;
    for (int i = tid; i < TILE * TILE; i += 256) {
        const int y = i >> 6, x = i & 63;
        const float v = src[(ys + y) * W_IMG + (xs + x)];
        mySq += v * v;
        myOv += (fabsf(v) > 0.f) ? 1 : 0;
        sS[y][x] = (_Float16)v;
    }
    for (int i = tid; i < SS_STRIDE; i += 256) sS[TILE][i] = (_Float16)0.f;
    {
        _Float16* wflat = &sW[0][0];
        for (int i = tid; i < WIN * SW_STRIDE; i += 256) wflat[i] = (_Float16)0.f;
        float* cflat = &sCorr[0][0];
        for (int i = tid; i < DGRID * 20; i += 256) cflat[i] = 0.f;
    }
    __syncthreads();

    // ---- phase 2: load 80x80 target window with zero border fill ----
    for (int i = tid; i < WIN * WIN; i += 256) {
        const int wy = i / WIN, wx = i - wy * WIN;
        const int gy = ys - MD + wy;
        const int gx = xs - MD + wx;
        float v = 0.f;
        if (gy >= 0 && gy < H_IMG && gx >= 0 && gx < W_IMG)
            v = tgt[gy * W_IMG + gx];
        sW[wy][wx] = (_Float16)v;
    }
    sRedF[tid] = mySq;
    sRedI[tid] = myOv;
    __syncthreads();

    // ---- phase 3: block reduction for src_sq and overlap ----
    for (int s = 128; s > 0; s >>= 1) {
        if (tid < s) { sRedF[tid] += sRedF[tid + s]; sRedI[tid] += sRedI[tid + s]; }
        __syncthreads();
    }

    // ---- phase 4: summed-area table of w^2 (81x81, zero top row / left col) ----
    if (tid <= WIN) sSAT[0][tid] = 0.f;
    if (tid < WIN) {
        float run = 0.f;
        sSAT[tid + 1][0] = 0.f;
        for (int c = 0; c < WIN; ++c) {
            const float w = (float)sW[tid][c];
            run += w * w;
            sSAT[tid + 1][c + 1] = run;
        }
    }
    __syncthreads();
    if (tid <= WIN) {
        float run = 0.f;
        for (int r = 1; r <= WIN; ++r) {
            run += sSAT[r][tid];
            sSAT[r][tid] = run;
        }
    }
    __syncthreads();

    // ---- phase 5: cross-correlation via WMMA, displacement-major ----
    // Per window row j:  R_j[dy, dx] = sum_x S[j-dy, x] * W[j, x+dx]
    // GEMM: M = dy (rows of A are reversed src rows, OOB -> zero row),
    //       N = dx (Hankel of window row j), K = x = 64 (2 k-steps).
    // corr[dy][dx] = sum_j R_j[dy][dx]  ->  accumulate in the WMMA C operand
    // across all j handled by this wave; merge once at the end.
    const int lrow  = lane & 15;        // M-row of A / N-col of B owned by this lane
    const int khalf = (lane >> 4) & 1;  // lanes 16-31 hold the +8 K-half

    v8f acc[2][2];
#pragma unroll
    for (int mt = 0; mt < 2; ++mt)
#pragma unroll
        for (int nt = 0; nt < 2; ++nt) acc[mt][nt] = (v8f){};

    for (int t = 0; t < 10; ++t) {
        const int j = wave + 8 * t;                        // window row, 0..79

        // A fragments: A[dy, x] = S[j - dy, x], dy = mt*16 + lrow (band dy>=17 discarded later)
        v16h a[2][2];
#pragma unroll
        for (int mt = 0; mt < 2; ++mt) {
            int row = j - (mt * 16 + lrow);
            if (row < 0 || row >= TILE) row = TILE;        // zero row
#pragma unroll
            for (int kb = 0; kb < 2; ++kb)
#pragma unroll
                for (int r = 0; r < 8; ++r) {
                    const int k = kb * 32 + ((r & 4) ? 16 : 0) + khalf * 8 + (r & 3) * 2;
                    a[mt][kb][2 * r]     = sS[row][k];
                    a[mt][kb][2 * r + 1] = sS[row][k + 1];
                }
        }

        // B fragments: B[x, dx] = W[j, x + dx], dx = nt*16 + lrow
        v16h b[2][2];
#pragma unroll
        for (int nt = 0; nt < 2; ++nt) {
            const int col = nt * 16 + lrow;
#pragma unroll
            for (int kb = 0; kb < 2; ++kb)
#pragma unroll
                for (int r = 0; r < 8; ++r) {
                    const int k = kb * 32 + ((r & 4) ? 16 : 0) + khalf * 8 + (r & 3) * 2;
                    b[nt][kb][2 * r]     = sW[j][k + col];
                    b[nt][kb][2 * r + 1] = sW[j][k + col + 1];
                }
        }

#pragma unroll
        for (int mt = 0; mt < 2; ++mt)
#pragma unroll
            for (int nt = 0; nt < 2; ++nt) {
                acc[mt][nt] = __builtin_amdgcn_wmma_f32_16x16x32_f16(
                        false, a[mt][0], false, b[nt][0], (short)0, acc[mt][nt], false, false);
                acc[mt][nt] = __builtin_amdgcn_wmma_f32_16x16x32_f16(
                        false, a[mt][1], false, b[nt][1], (short)0, acc[mt][nt], false, false);
            }
    }

    // one-shot merge of this wave's partial corr into shared corr
#pragma unroll
    for (int mt = 0; mt < 2; ++mt)
#pragma unroll
        for (int nt = 0; nt < 2; ++nt)
#pragma unroll
            for (int e = 0; e < 8; ++e) {
                const int dy = mt * 16 + e + khalf * 8;
                const int dx = nt * 16 + lrow;
                if (dy < DGRID && dx < DGRID)
                    atomicAdd(&sCorr[dy][dx], acc[mt][nt][e]);   // ds_add_f32
            }
    __syncthreads();

    // ---- phase 6: SSD costs for all 289 displacements ----
    if (tid < DGRID * DGRID) {
        const int dy = tid / DGRID, dx = tid - dy * DGRID;
        const float wsq = sSAT[dy + TILE][dx + TILE] - sSAT[dy][dx + TILE]
                        - sSAT[dy + TILE][dx]        + sSAT[dy][dx];
        sCost[tid] = sRedF[0] + wsq - 2.0f * sCorr[dy][dx];
    }
    __syncthreads();

    // ---- phase 7: argmin + second-best, peak test, emit (dx,dy) ----
    if (tid == 0) {
        float c0 = 3.4e38f, c1 = 3.4e38f;
        int idx = 0;
        for (int i = 0; i < DGRID * DGRID; ++i) {
            const float c = sCost[i];
            if (c < c0)      { c1 = c0; c0 = c; idx = i; }
            else if (c < c1) { c1 = c; }
        }
        const bool ok = (c1 / (c0 + 1e-6f) >= 1.1f) && (sRedI[0] >= 500);
        const float fdx = ok ? (float)(idx % DGRID - MD) : 0.f;
        const float fdy = ok ? (float)(idx / DGRID - MD) : 0.f;
        const int tl = ty * TGRID + tx;
        tile_out[2 * tl]     = fdx;
        tile_out[2 * tl + 1] = fdy;
    }
}

// Nearest-neighbor upsample of the 63x63 (dx,dy) field to [H,W,2].
// JAX nearest: in_idx = floor((out_idx + 0.5) * in/out).
__global__ __launch_bounds__(256)
void bm_upsample_kernel(const float* __restrict__ tile_out,
                        float2* __restrict__ out)
{
    const int i = blockIdx.x * blockDim.x + threadIdx.x;
    if (i >= H_IMG * W_IMG) return;
    const int h = i / W_IMG;
    const int w = i - h * W_IMG;
    const int tyi = ((2 * h + 1) * TGRID) / (2 * H_IMG);
    const int txi = ((2 * w + 1) * TGRID) / (2 * W_IMG);
    const int tl  = tyi * TGRID + txi;
    out[i] = make_float2(tile_out[2 * tl], tile_out[2 * tl + 1]);
}

extern "C" void kernel_launch(void* const* d_in, const int* in_sizes, int n_in,
                              void* d_out, int out_size, void* d_ws, size_t ws_size,
                              hipStream_t stream) {
    (void)in_sizes; (void)n_in; (void)out_size; (void)ws_size;
    const float* src = (const float*)d_in[0];
    const float* tgt = (const float*)d_in[1];
    float* tile_ws = (float*)d_ws;                 // TGRID*TGRID*2 floats

    dim3 grid1(TGRID, TGRID);
    bm_cost_kernel<<<grid1, dim3(256), 0, stream>>>(src, tgt, tile_ws);

    const int npix = H_IMG * W_IMG;
    bm_upsample_kernel<<<dim3((npix + 255) / 256), dim3(256), 0, stream>>>(
        tile_ws, (float2*)d_out);
}
